// AttentionOpt_9423158247824
// MI455X (gfx1250) — compile-verified, matrix operand order checked by reading, not run on hardware
//
#include <hip/hip_runtime.h>

// ---------------------------------------------------------------------------
// MI455X (gfx1250) fused masked-attention + layernorm.
// All matmuls in bf16 via v_wmma_f32_16x16x32_bf16 (f32 accumulation).
// K-tiles staged via global_load_async_to_lds_b128 (ASYNCcnt path).
// ---------------------------------------------------------------------------

typedef __attribute__((ext_vector_type(16))) __bf16 v16bf;
typedef __attribute__((ext_vector_type(8)))  __bf16 v8bf;
typedef __attribute__((ext_vector_type(8)))  float  v8f;

#define N_B   8
#define C_DIM 1024
#define L_DIM 1024
#define H_N   16
#define DH    64
#define NEGF  (-4294967295.0f)   // -(2^32 - 1), matches reference NEG

#define USE_ASYNC_LDS 1

// gfx1250 has native bf16: let the backend emit the hardware f32->bf16 cvt
// instead of a 3-op software round-to-nearest (big VALU saving in k_qkv).
static __device__ __forceinline__ __bf16 f2bf(float f) {
  return (__bf16)f;
}

// Build a 16-element bf16 fragment from a contiguous 32-element K-span.
// Per CDNA5 ISA 16-bit A/B layouts: lanes 0-15 take K=[0,8)+[16,24),
// lanes 16-31 take K=[8,16)+[24,32)  (half = lane>>4).
static __device__ __forceinline__ v16bf ldfrag(const __bf16* p, int half) {
  v8bf lo = *(const v8bf*)(p + half * 8);
  v8bf hi = *(const v8bf*)(p + 16 + half * 8);
  v16bf r;
#pragma unroll
  for (int i = 0; i < 8; ++i) { r[i] = lo[i]; r[i + 8] = hi[i]; }
  return r;
}

// ---------------------------------------------------------------------------
// Kernel 1: Y[n,l,d] = relu(sum_c seq[n,c,l] * W[d,c] + b[d])  -> bf16
// stored as [n, h, l, dh].  Block: 128(l) x 64(d) tile, 8 waves, each wave a
// 16x64 strip (4 WMMA accumulators).  K-loop over C in steps of 32.
// ---------------------------------------------------------------------------
__global__ __launch_bounds__(256) void k_qkv(
    const float* __restrict__ seq,
    const float* __restrict__ wq, const float* __restrict__ bq,
    const float* __restrict__ wk, const float* __restrict__ bk,
    const float* __restrict__ wv, const float* __restrict__ bv,
    __bf16* __restrict__ Qb, __bf16* __restrict__ Kb, __bf16* __restrict__ Vb)
{
  const int lt = blockIdx.x;            // l tile (128 rows)
  const int dt = blockIdx.y;            // d tile (64 cols == one head)
  const int n  = blockIdx.z / 3;
  const int which = blockIdx.z % 3;
  const float* W; const float* bias; __bf16* dst;
  if (which == 0)      { W = wq; bias = bq; dst = Qb; }
  else if (which == 1) { W = wk; bias = bk; dst = Kb; }
  else                 { W = wv; bias = bv; dst = Vb; }

  __shared__ __attribute__((aligned(16))) __bf16 As[128 * 32];  // [l][c]
  __shared__ __attribute__((aligned(16))) __bf16 Bs[64 * 32];   // [d][c]

  const int t    = threadIdx.x;
  const int lane = t & 31, wave = t >> 5;
  const int half = lane >> 4, l16 = lane & 15;

  v8f acc[4];
#pragma unroll
  for (int j = 0; j < 4; ++j)
#pragma unroll
    for (int e = 0; e < 8; ++e) acc[j][e] = 0.f;

  const float* seqn = seq + (size_t)n * C_DIM * L_DIM;
  const int cl = t >> 5;                // 0..7   (c sub-index for A loads)
  const int lq = (t & 31) << 2;         // 0..124 (l offset, float4 granules)
  const int dl = t >> 3;                // 0..31  (d sub-index for B loads)
  const int cq = (t & 7) << 2;          // 0..28  (c offset, float4 granules)

  for (int c0 = 0; c0 < C_DIM; c0 += 32) {
    // A tile: x[l,c] = seq[n,c,l]; coalesced along l, convert to bf16.
#pragma unroll
    for (int j = 0; j < 4; ++j) {
      const int cc = cl + j * 8;
      const float4 v =
          *(const float4*)(seqn + (size_t)(c0 + cc) * L_DIM + lt * 128 + lq);
      As[(lq + 0) * 32 + cc] = f2bf(v.x);
      As[(lq + 1) * 32 + cc] = f2bf(v.y);
      As[(lq + 2) * 32 + cc] = f2bf(v.z);
      As[(lq + 3) * 32 + cc] = f2bf(v.w);
    }
    // B tile: W[d,c]; coalesced along c.
#pragma unroll
    for (int j = 0; j < 2; ++j) {
      const int dd = dl + j * 32;
      const float4 v =
          *(const float4*)(W + (size_t)(dt * 64 + dd) * C_DIM + c0 + cq);
      __bf16* bp = &Bs[dd * 32 + cq];
      bp[0] = f2bf(v.x); bp[1] = f2bf(v.y); bp[2] = f2bf(v.z); bp[3] = f2bf(v.w);
    }
    __syncthreads();

    const v16bf af = ldfrag(&As[(wave * 16 + l16) * 32], half);
#pragma unroll
    for (int j = 0; j < 4; ++j) {
      const v16bf bfp = ldfrag(&Bs[(j * 16 + l16) * 32], half);
      acc[j] = __builtin_amdgcn_wmma_f32_16x16x32_bf16(
          false, af, false, bfp, (short)0, acc[j], false, false);
    }
    __syncthreads();
  }

  // Epilogue: bias + relu, store bf16 as [n, h, l, dh].
#pragma unroll
  for (int j = 0; j < 4; ++j) {
    const int dg   = dt * 64 + j * 16 + l16;
    const float bv2 = bias[dg];
    const int hh = dg >> 6, di = dg & 63;
#pragma unroll
    for (int i = 0; i < 8; ++i) {
      const int lg = lt * 128 + wave * 16 + i + half * 8;  // C/D frag row map
      float y = acc[j][i] + bv2;
      y = y > 0.f ? y : 0.f;
      dst[((size_t)(n * H_N + hh) * L_DIM + lg) * DH + di] = f2bf(y);
    }
  }
}

// ---------------------------------------------------------------------------
// Kernel 2: flash-attention per (n, h, 128 q-rows).  Wave = 16 q-rows.
// Online softmax; key blocks of 32.  S = Q K^T via 4 WMMA, O += P V via 4.
// K tile staged with CDNA5 async global->LDS DMA (ASYNCcnt).
// ---------------------------------------------------------------------------
__global__ __launch_bounds__(256) void k_attn(
    const __bf16* __restrict__ Qb, const __bf16* __restrict__ Kb,
    const __bf16* __restrict__ Vb, const int* __restrict__ mask,
    float* __restrict__ obuf)
{
  const int qt = blockIdx.x, h = blockIdx.y, n = blockIdx.z;
  const int t = threadIdx.x, lane = t & 31, wave = t >> 5;
  const int half = lane >> 4, l16 = lane & 15;

  __shared__ __attribute__((aligned(16))) int    msk[L_DIM];
  __shared__ __attribute__((aligned(16))) __bf16 Ks[32 * 64];      // [k][d]
  __shared__ __attribute__((aligned(16))) __bf16 VsT[64 * 32];     // [d][k]
  __shared__ __attribute__((aligned(16))) __bf16 Ps[8][16 * 32];   // per wave

  for (int i = t; i < L_DIM; i += 256) msk[i] = mask[n * L_DIM + i];

  const size_t bh = ((size_t)(n * H_N + h)) * L_DIM * DH;
  const __bf16* Qh = Qb + bh;
  const __bf16* Kh = Kb + bh;
  const __bf16* Vh = Vb + bh;

  // Q A-fragments for this wave's 16 rows (d = 0..31 and 32..63).
  const int qrow = qt * 128 + wave * 16 + l16;
  const v16bf qf0 = ldfrag(Qh + (size_t)qrow * DH, half);
  const v16bf qf1 = ldfrag(Qh + (size_t)qrow * DH + 32, half);

  float mrow[8], lrow[8];
  v8f O[4];
#pragma unroll
  for (int i = 0; i < 8; ++i) { mrow[i] = -__builtin_inff(); lrow[i] = 0.f; }
#pragma unroll
  for (int j = 0; j < 4; ++j)
#pragma unroll
    for (int e = 0; e < 8; ++e) O[j][e] = 0.f;

  const int kl = t >> 3;            // 0..31 key row for cooperative loads
  const int dq = (t & 7) << 3;      // 0..56 d offset (8 bf16 = 16 B)

  __syncthreads();                  // msk ready

  for (int kb = 0; kb < L_DIM; kb += 32) {
    // --- Stage K tile [32][64]: async DMA straight into LDS (no VGPR hop).
#if USE_ASYNC_LDS
    {
      const unsigned lds_dst = (unsigned)(uintptr_t)(&Ks[kl * 64 + dq]);
      const unsigned long long gsrc =
          (unsigned long long)(uintptr_t)(Kh + (size_t)(kb + kl) * DH + dq);
      asm volatile("global_load_async_to_lds_b128 %0, %1, off"
                   :: "v"(lds_dst), "v"(gsrc) : "memory");
    }
#else
    *(v8bf*)&Ks[kl * 64 + dq] = *(const v8bf*)(Kh + (size_t)(kb + kl) * DH + dq);
#endif
    // --- Stage transposed V tile [64][32] (needs element transpose).
    const v8bf vv = *(const v8bf*)(Vh + (size_t)(kb + kl) * DH + dq);
#pragma unroll
    for (int e = 0; e < 8; ++e) VsT[(dq + e) * 32 + kl] = vv[e];
    if (kb + 32 < L_DIM) {
      __builtin_prefetch(Kh + (size_t)(kb + 32 + kl) * DH + dq, 0, 1);
      __builtin_prefetch(Vh + (size_t)(kb + 32 + kl) * DH + dq, 0, 1);
    }
#if USE_ASYNC_LDS
    asm volatile("s_wait_asynccnt 0x0" ::: "memory");  // our K DMA done
#endif
    __syncthreads();

    // S = Q K^T : two 16-key column tiles, 2 WMMA each over d.
    v8f s[2];
#pragma unroll
    for (int tt = 0; tt < 2; ++tt) {
      v8f a;
#pragma unroll
      for (int e = 0; e < 8; ++e) a[e] = 0.f;
      a = __builtin_amdgcn_wmma_f32_16x16x32_bf16(
          false, qf0, false, ldfrag(&Ks[(tt * 16 + l16) * 64], half),
          (short)0, a, false, false);
      a = __builtin_amdgcn_wmma_f32_16x16x32_bf16(
          false, qf1, false, ldfrag(&Ks[(tt * 16 + l16) * 64 + 32], half),
          (short)0, a, false, false);
      s[tt] = a;
    }
    // Scale + key mask: fragment column is uniform per lane.
#pragma unroll
    for (int tt = 0; tt < 2; ++tt) {
      const int masked = (msk[kb + tt * 16 + l16] == 0);
#pragma unroll
      for (int i = 0; i < 8; ++i)
        s[tt][i] = masked ? NEGF : s[tt][i] * 0.125f;
    }
    // Row max over 16 lanes of each half (xor 1,2,4,8 stays inside a half).
    float mloc[8];
#pragma unroll
    for (int i = 0; i < 8; ++i) mloc[i] = fmaxf(s[0][i], s[1][i]);
#pragma unroll
    for (int off = 1; off < 16; off <<= 1)
#pragma unroll
      for (int i = 0; i < 8; ++i)
        mloc[i] = fmaxf(mloc[i], __shfl_xor(mloc[i], off, 32));
    float sc[8], rs[8];
#pragma unroll
    for (int i = 0; i < 8; ++i) {
      const float mn = fmaxf(mrow[i], mloc[i]);
      sc[i] = __expf(mrow[i] - mn);
      mrow[i] = mn;
      s[0][i] = __expf(s[0][i] - mn);
      s[1][i] = __expf(s[1][i] - mn);
      rs[i] = s[0][i] + s[1][i];
    }
#pragma unroll
    for (int off = 1; off < 16; off <<= 1)
#pragma unroll
      for (int i = 0; i < 8; ++i) rs[i] += __shfl_xor(rs[i], off, 32);
#pragma unroll
    for (int i = 0; i < 8; ++i) lrow[i] = lrow[i] * sc[i] + rs[i];
#pragma unroll
    for (int j = 0; j < 4; ++j)
#pragma unroll
      for (int i = 0; i < 8; ++i) O[j][i] *= sc[i];

    // Re-layout P: C-fragment -> A-fragment via per-wave LDS scratch.
    __bf16* pw = &Ps[wave][0];
#pragma unroll
    for (int tt = 0; tt < 2; ++tt)
#pragma unroll
      for (int i = 0; i < 8; ++i)
        pw[(i + half * 8) * 32 + tt * 16 + l16] = f2bf(s[tt][i]);
    asm volatile("s_wait_dscnt 0x0" ::: "memory");   // same-wave DS RAW fence
    const v16bf pf = ldfrag(pw + l16 * 32, half);

    // O += P V
#pragma unroll
    for (int j = 0; j < 4; ++j) {
      const v16bf vf = ldfrag(&VsT[(j * 16 + l16) * 32], half);
      O[j] = __builtin_amdgcn_wmma_f32_16x16x32_bf16(
          false, pf, false, vf, (short)0, O[j], false, false);
    }
    __syncthreads();   // protect Ks/VsT before next iteration's stores
  }

  // Epilogue: normalize, apply query mask, store f32 as [n, l, c].
  float linv[8], qm[8];
#pragma unroll
  for (int i = 0; i < 8; ++i) {
    linv[i] = lrow[i] > 0.f ? 1.f / lrow[i] : 0.f;
    qm[i]   = msk[qt * 128 + wave * 16 + i + half * 8] ? 1.f : 0.f;
  }
#pragma unroll
  for (int j = 0; j < 4; ++j) {
    const int cg = h * DH + j * 16 + l16;
#pragma unroll
    for (int i = 0; i < 8; ++i) {
      const int qg = qt * 128 + wave * 16 + i + half * 8;
      obuf[((size_t)n * L_DIM + qg) * C_DIM + cg] = O[j][i] * linv[i] * qm[i];
    }
  }
}

// ---------------------------------------------------------------------------
// Kernel 3: channel LayerNorm per (n, l) + mask; output [n, c, l] f32.
// One wave per row; wave-shuffle reduction.
// ---------------------------------------------------------------------------
__global__ __launch_bounds__(256) void k_ln(
    const float* __restrict__ obuf, const int* __restrict__ mask,
    const float* __restrict__ lnw, const float* __restrict__ lnb,
    float* __restrict__ out)
{
  const int t = threadIdx.x, lane = t & 31, wave = t >> 5;
  const int row = blockIdx.x * 8 + wave;     // 0 .. N*L-1
  const int n = row >> 10, l = row & 1023;
  const float* y = obuf + (size_t)row * C_DIM;

  float v[32];
  float sum = 0.f, sq = 0.f;
#pragma unroll
  for (int j = 0; j < 8; ++j) {
    const float4 f = *(const float4*)(y + j * 128 + lane * 4);
    v[j * 4 + 0] = f.x; v[j * 4 + 1] = f.y; v[j * 4 + 2] = f.z; v[j * 4 + 3] = f.w;
    sum += f.x + f.y + f.z + f.w;
    sq  += f.x * f.x + f.y * f.y + f.z * f.z + f.w * f.w;
  }
#pragma unroll
  for (int off = 16; off >= 1; off >>= 1) {
    sum += __shfl_xor(sum, off, 32);
    sq  += __shfl_xor(sq,  off, 32);
  }
  const float mean = sum * (1.f / 1024.f);
  const float var  = sq * (1.f / 1024.f) - mean * mean;
  const float inv  = rsqrtf(var + 1e-5f);
  const float mm   = mask[n * L_DIM + l] ? 1.f : 0.f;
#pragma unroll
  for (int j = 0; j < 8; ++j)
#pragma unroll
    for (int e = 0; e < 4; ++e) {
      const int c = j * 128 + lane * 4 + e;
      out[((size_t)n * C_DIM + c) * L_DIM + l] =
          (lnw[c] * ((v[j * 4 + e] - mean) * inv) + lnb[c]) * mm;
    }
}

// ---------------------------------------------------------------------------
extern "C" void kernel_launch(void* const* d_in, const int* in_sizes, int n_in,
                              void* d_out, int out_size, void* d_ws, size_t ws_size,
                              hipStream_t stream) {
  const float* seq = (const float*)d_in[0];
  const int*   msk = (const int*)  d_in[1];
  const float* wq  = (const float*)d_in[2];
  const float* bq  = (const float*)d_in[3];
  const float* wk  = (const float*)d_in[4];
  const float* bk  = (const float*)d_in[5];
  const float* wv  = (const float*)d_in[6];
  const float* bv  = (const float*)d_in[7];
  const float* lnw = (const float*)d_in[8];
  const float* lnb = (const float*)d_in[9];

  // Workspace: 3 x 16 MB bf16 QKV + 32 MB f32 attention output = 80 MB.
  char* ws = (char*)d_ws;
  const size_t qkv_bytes = (size_t)N_B * L_DIM * C_DIM * sizeof(__bf16);
  __bf16* Qb = (__bf16*)(ws);
  __bf16* Kb = (__bf16*)(ws + qkv_bytes);
  __bf16* Vb = (__bf16*)(ws + 2 * qkv_bytes);
  float*  ob = (float*) (ws + 3 * qkv_bytes);

  dim3 g1(L_DIM / 128, C_DIM / 64, N_B * 3);
  k_qkv<<<g1, dim3(256), 0, stream>>>(seq, wq, bq, wk, bk, wv, bv, Qb, Kb, Vb);

  dim3 g2(L_DIM / 128, H_N, N_B);
  k_attn<<<g2, dim3(256), 0, stream>>>(Qb, Kb, Vb, msk, ob);

  k_ln<<<dim3((N_B * L_DIM) / 8), dim3(256), 0, stream>>>(ob, msk, lnw, lnb,
                                                          (float*)d_out);
}
